// LaCTAudioBlock_12670153523859
// MI455X (gfx1250) — matrix operand order repaired and checked by reading, not compiled, not run on hardware
//
#include <hip/hip_runtime.h>
#include <hip/hip_bf16.h>
#include <string.h>

// ---------------------------------------------------------------------------
// LaCT audio block for MI455X (gfx1250): WMMA f16->f32 + async global->LDS.
// ---------------------------------------------------------------------------

typedef __attribute__((ext_vector_type(16))) _Float16 v16h;
typedef __attribute__((ext_vector_type(8)))  _Float16 v8h;
typedef __attribute__((ext_vector_type(8)))  float    v8f;

#define HS   1024
#define SQ   4096
#define NHA  16
#define HD   64
#define NHL  4
#define DIN  256
#define CHK  1024
#define NC   4
#define NZ   16          // NC * NHL
#define PERZ (CHK * DIN) // elements per (chunk,head) activation slab

static __device__ __forceinline__ v8f wmma_f16(v16h a, v16h b, v8f c) {
  return __builtin_amdgcn_wmma_f32_16x16x32_f16(false, a, false, b, (short)0, c,
                                                false, false);
}

// ---- fragment loaders (contiguous k: lane half h holds k = 16h..16h+15) ----------------------
// A row-major [M x K]: rows m0..m0+15, cols k0..k0+31
static __device__ __forceinline__ v16h load_a_rm(const _Float16* A, long lda,
                                                 int m0, int k0, int lane) {
  long base = (long)(m0 + (lane & 15)) * lda + k0 + ((lane >> 4) << 4);
  v8h lo = *(const v8h*)(A + base);
  v8h hi = *(const v8h*)(A + base + 8);
  v16h r;
#pragma unroll
  for (int j = 0; j < 8; ++j) { r[j] = lo[j]; r[j + 8] = hi[j]; }
  return r;
}
// A transposed: memory is [K x M] row-major; element (m,k) at mem[k*lda+m]
static __device__ __forceinline__ v16h load_a_tr(const _Float16* A, long lda,
                                                 int m0, int k0, int lane) {
  int m  = m0 + (lane & 15);
  int kb = k0 + ((lane >> 4) << 4);
  v16h r;
#pragma unroll
  for (int j = 0; j < 8; ++j) {
    r[j]     = A[(long)(kb + j) * lda + m];
    r[j + 8] = A[(long)(kb + 8 + j) * lda + m];
  }
  return r;
}
// B row-major [K x N]
static __device__ __forceinline__ v16h load_b_rm(const _Float16* B, long ldb,
                                                 int k0, int n0, int lane) {
  int n  = n0 + (lane & 15);
  int kb = k0 + ((lane >> 4) << 4);
  v16h r;
#pragma unroll
  for (int j = 0; j < 16; ++j) r[j] = B[(long)(kb + j) * ldb + n];
  return r;
}
// B transposed: memory is [N x K] row-major; element (k,n) at mem[n*ldb+k]
static __device__ __forceinline__ v16h load_b_tr(const _Float16* B, long ldb,
                                                 int k0, int n0, int lane) {
  long base = (long)(n0 + (lane & 15)) * ldb + k0 + ((lane >> 4) << 4);
  v8h lo = *(const v8h*)(B + base);
  v8h hi = *(const v8h*)(B + base + 8);
  v16h r;
#pragma unroll
  for (int j = 0; j < 8; ++j) { r[j] = lo[j]; r[j + 8] = hi[j]; }
  return r;
}

// ---- async global->LDS tile staging (CDNA5 ASYNCcnt path) -----------------
// Copies ROWS x COLS(_Float16) with global row stride ldg into a dense LDS
// tile (row stride COLS). 16B chunks per lane via global_load_async_to_lds.
template <int ROWS, int COLS, int NT>
static __device__ __forceinline__ void stage_async(const _Float16* g, long ldg,
                                                   _Float16* l, int tid) {
  constexpr int CPR    = COLS / 8;     // 16B chunks per row
  constexpr int CHUNKS = ROWS * CPR;
#pragma unroll
  for (int c = tid; c < CHUNKS; c += NT) {
    int row = c / CPR;
    int col = (c % CPR) * 8;
    const _Float16* src = g + (long)row * ldg + col;
    unsigned dst = (unsigned)(unsigned long long)(l + row * COLS + col);
    asm volatile("global_load_async_to_lds_b128 %0, %1, off"
                 :
                 : "v"(dst), "v"((unsigned long long)src)
                 : "memory");
  }
}
static __device__ __forceinline__ void wait_async0() {
  asm volatile("s_wait_asynccnt 0" ::: "memory");
}
static __device__ __forceinline__ void wait_ds0() {
  asm volatile("s_wait_dscnt 0" ::: "memory");
}

// ---------------------------------------------------------------------------
// Generic batched WMMA GEMM. Block: 128 thr = 4 waves. Block tile 64x64,
// wave tile 64x16. A/B K-slices (64x32 tiles) double-buffered through LDS
// via async copies; fragments then come from LDS with compile-time strides.
// ---------------------------------------------------------------------------
struct GemmP {
  const _Float16* A; long lda, Azn, Azh;
  const _Float16* B; long ldb, Bzn, Bzh;
  const float*    C; long ldc, Czn, Czh;  // optional f32 accumulator init
  const float* bias;                      // optional per-output-column bias
  float*    Df; long lddf, Dfzn, Dfzh; int accum;
  _Float16* Dh; long lddh, Dhzn, Dhzh;
  int M, N, K, zdiv;
};

template <bool AT, bool BT>
__global__ __launch_bounds__(128) void gemm_wmma(GemmP p) {
  __shared__ __align__(16) _Float16 As[2][64 * 32];
  __shared__ __align__(16) _Float16 Bs[2][64 * 32];

  int zn = blockIdx.z / p.zdiv, zh = blockIdx.z % p.zdiv;
  const _Float16* A = p.A + (long)zn * p.Azn + (long)zh * p.Azh;
  const _Float16* B = p.B + (long)zn * p.Bzn + (long)zh * p.Bzh;
  int tid  = threadIdx.x;
  int lane = tid & 31;
  int wv   = tid >> 5;
  int m0   = blockIdx.x * 64;
  int n0   = blockIdx.y * 64;
  int nw   = wv * 16;              // wave's 16-col slice within the 64 tile
  int rb   = (lane >> 4) << 3;
  int cn   = n0 + nw + (lane & 15);

  // global base of the (m0,n0) block's K-slice at k0:
  auto Aslice = [&](int k0) {
    return AT ? (A + (long)k0 * p.lda + m0) : (A + (long)m0 * p.lda + k0);
  };
  auto Bslice = [&](int k0) {
    return BT ? (B + (long)(n0) * p.ldb + k0) : (B + (long)k0 * p.ldb + n0);
  };
  auto stageA = [&](int k0, int buf) {
    if constexpr (AT) stage_async<32, 64, 128>(Aslice(k0), p.lda, As[buf], tid);
    else              stage_async<64, 32, 128>(Aslice(k0), p.lda, As[buf], tid);
  };
  auto stageB = [&](int k0, int buf) {
    if constexpr (BT) stage_async<64, 32, 128>(Bslice(k0), p.ldb, Bs[buf], tid);
    else              stage_async<32, 64, 128>(Bslice(k0), p.ldb, Bs[buf], tid);
  };

  v8f acc[4];
  if (p.C) {
    const float* C = p.C + (long)zn * p.Czn + (long)zh * p.Czh;
#pragma unroll
    for (int t = 0; t < 4; ++t)
#pragma unroll
      for (int r = 0; r < 8; ++r)
        acc[t][r] = C[(long)(m0 + t * 16 + rb + r) * p.ldc + cn];
  } else {
#pragma unroll
    for (int t = 0; t < 4; ++t)
#pragma unroll
      for (int r = 0; r < 8; ++r) acc[t][r] = 0.f;
  }

  stageA(0, 0);
  stageB(0, 0);
  int buf = 0;
  for (int k0 = 0; k0 < p.K; k0 += 32) {
    wait_async0();          // this wave's copies for `buf` are in LDS
    __syncthreads();        // ... and so are everyone else's
    if (k0 + 32 < p.K) {    // overlap next K-slice copy with this compute
      stageA(k0 + 32, buf ^ 1);
      stageB(k0 + 32, buf ^ 1);
    }
    if (k0 + 64 < p.K) {    // nudge the slice after that toward L2
      __builtin_prefetch(Aslice(k0 + 64) + tid * 8, 0, 1);
      __builtin_prefetch(Bslice(k0 + 64) + tid * 8, 0, 1);
    }
    v16h b;
    if constexpr (BT) b = load_b_tr(Bs[buf], 32, 0, nw, lane);
    else              b = load_b_rm(Bs[buf], 64, 0, nw, lane);
#pragma unroll
    for (int t = 0; t < 4; ++t) {
      v16h a;
      if constexpr (AT) a = load_a_tr(As[buf], 64, t * 16, 0, lane);
      else              a = load_a_rm(As[buf], 32, t * 16, 0, lane);
      acc[t] = wmma_f16(a, b, acc[t]);
    }
    buf ^= 1;
  }

  float bv = p.bias ? p.bias[cn] : 0.f;
  if (p.Df) {
    float* Df = p.Df + (long)zn * p.Dfzn + (long)zh * p.Dfzh;
#pragma unroll
    for (int t = 0; t < 4; ++t)
#pragma unroll
      for (int r = 0; r < 8; ++r) {
        long idx = (long)(m0 + t * 16 + rb + r) * p.lddf + cn;
        float v  = acc[t][r] + bv;
        Df[idx]  = p.accum ? (Df[idx] + v) : v;
      }
  }
  if (p.Dh) {
    _Float16* Dh = p.Dh + (long)zn * p.Dhzn + (long)zh * p.Dhzh;
#pragma unroll
    for (int t = 0; t < 4; ++t)
#pragma unroll
      for (int r = 0; r < 8; ++r)
        Dh[(long)(m0 + t * 16 + rb + r) * p.lddh + cn] =
            (_Float16)(acc[t][r] + bv);
  }
}

// ---------------------------------------------------------------------------
// LayerNorm: one row (1024 cols) per block, 256 threads. Output f16.
// ---------------------------------------------------------------------------
__global__ __launch_bounds__(256) void ln_kernel(const float* __restrict__ x,
                                                 const float* __restrict__ w,
                                                 const float* __restrict__ b,
                                                 _Float16* __restrict__ xnh) {
  long row = blockIdx.x;
  const float* xr = x + row * HS;
  float s = 0.f, ss = 0.f;
  for (int i = threadIdx.x; i < HS; i += 256) {
    float v = xr[i];
    s += v; ss += v * v;
  }
#pragma unroll
  for (int off = 16; off > 0; off >>= 1) {
    s  += __shfl_xor(s, off, 32);
    ss += __shfl_xor(ss, off, 32);
  }
  __shared__ float sh[2][8];
  int lane = threadIdx.x & 31, wv = threadIdx.x >> 5;
  if (lane == 0) { sh[0][wv] = s; sh[1][wv] = ss; }
  __syncthreads();
  float ts = 0.f, tss = 0.f;
#pragma unroll
  for (int i = 0; i < 8; ++i) { ts += sh[0][i]; tss += sh[1][i]; }
  float mu  = ts * (1.f / HS);
  float var = tss * (1.f / HS) - mu * mu;
  float inv = rsqrtf(var + 1e-5f);
  for (int i = threadIdx.x; i < HS; i += 256)
    xnh[row * HS + i] = (_Float16)((xr[i] - mu) * inv * w[i] + b[i]);
}

// ---------------------------------------------------------------------------
// Flash attention: one wave per (head, 16-query-row tile). hd=64, scale=1/8.
// V tile async-staged into wave-private LDS overlapping QK^T + softmax;
// P tile staged D-layout -> LDS -> A-layout.
// ---------------------------------------------------------------------------
static __device__ __forceinline__ float rmax16(float v) {
#pragma unroll
  for (int m = 1; m < 16; m <<= 1) v = fmaxf(v, __shfl_xor(v, m, 16));
  return v;
}
static __device__ __forceinline__ float rsum16(float v) {
#pragma unroll
  for (int m = 1; m < 16; m <<= 1) v += __shfl_xor(v, m, 16);
  return v;
}

__global__ __launch_bounds__(128) void flash_attn(const _Float16* __restrict__ qkv,
                                                  _Float16* __restrict__ ao,
                                                  int S) {
  __shared__ __align__(16) _Float16 Pl[4][16][32];
  __shared__ __align__(16) _Float16 Vl[4][32][64];
  int lane = threadIdx.x & 31;
  int wv   = threadIdx.x >> 5;
  int gw   = blockIdx.x * 4 + wv;
  int head = gw >> 8;        // S/16 = 256 m-tiles per head
  int m0   = (gw & 255) * 16;
  const long ldq = 3 * HS;
  const _Float16* Q  = qkv + head * HD;
  const _Float16* Km = qkv + HS + head * HD;
  const _Float16* Vm = qkv + 2 * HS + head * HD;

  v16h aq0 = load_a_rm(Q, ldq, m0, 0, lane);
  v16h aq1 = load_a_rm(Q, ldq, m0, 32, lane);

  v8f o[4];
#pragma unroll
  for (int t = 0; t < 4; ++t)
#pragma unroll
    for (int r = 0; r < 8; ++r) o[t][r] = 0.f;
  float mrow[8], lrow[8];
#pragma unroll
  for (int r = 0; r < 8; ++r) { mrow[r] = -3.0e38f; lrow[r] = 0.f; }

  int rb = (lane >> 4) << 3;
  int cx = lane & 15;

  for (int kb = 0; kb < S; kb += 32) {
    // prior iteration's LDS reads (Pl, Vl) must retire before restaging
    wait_ds0();
    stage_async<32, 64, 32>(Vm + (long)kb * ldq, ldq, &Vl[wv][0][0], lane);

    v8f s0, s1;
#pragma unroll
    for (int r = 0; r < 8; ++r) { s0[r] = 0.f; s1[r] = 0.f; }
#pragma unroll
    for (int kk = 0; kk < 2; ++kk) {
      v16h aq = kk ? aq1 : aq0;
      v16h b0 = load_b_tr(Km, ldq, kk * 32, kb, lane);
      v16h b1 = load_b_tr(Km, ldq, kk * 32, kb + 16, lane);
      s0 = wmma_f16(aq, b0, s0);
      s1 = wmma_f16(aq, b1, s1);
    }
    const float sc = 0.125f;  // 1/sqrt(64)
    float p0[8], p1[8], alpha[8];
#pragma unroll
    for (int r = 0; r < 8; ++r) {
      float v0 = s0[r] * sc, v1 = s1[r] * sc;
      float mx = rmax16(fmaxf(v0, v1));
      float mn = fmaxf(mrow[r], mx);
      float a  = __expf(mrow[r] - mn);
      float e0 = __expf(v0 - mn);
      float e1 = __expf(v1 - mn);
      float rs = rsum16(e0 + e1);
      lrow[r]  = lrow[r] * a + rs;
      mrow[r]  = mn;
      alpha[r] = a;
      p0[r] = e0; p1[r] = e1;
    }
#pragma unroll
    for (int t = 0; t < 4; ++t)
#pragma unroll
      for (int r = 0; r < 8; ++r) o[t][r] *= alpha[r];

    // D-layout -> LDS (row-major 16x32), then reload as A fragment.
#pragma unroll
    for (int r = 0; r < 8; ++r) {
      Pl[wv][rb + r][cx]      = (_Float16)p0[r];
      Pl[wv][rb + r][cx + 16] = (_Float16)p1[r];
    }
    wait_ds0();
    v16h ap = load_a_rm(&Pl[wv][0][0], 32, 0, 0, lane);

    wait_async0();  // V tile resident
#pragma unroll
    for (int t = 0; t < 4; ++t) {
      v16h bv = load_b_rm(&Vl[wv][0][0], 64, 0, t * 16, lane);
      o[t] = wmma_f16(ap, bv, o[t]);
    }
  }

#pragma unroll
  for (int r = 0; r < 8; ++r) lrow[r] = 1.f / lrow[r];
#pragma unroll
  for (int t = 0; t < 4; ++t)
#pragma unroll
    for (int r = 0; r < 8; ++r)
      ao[(long)(m0 + rb + r) * HS + head * HD + t * 16 + cx] =
          (_Float16)(o[t][r] * lrow[r]);
}

// ---------------------------------------------------------------------------
// Elementwise kernels (SwiGLU forward + gradient algebra, signs folded so
// every weight update becomes  w_new = w + A^T B ).
// ---------------------------------------------------------------------------
__global__ void lact_ew1(const float* __restrict__ gate, const float* __restrict__ g,
                         const float* __restrict__ dhp, _Float16* __restrict__ hid,
                         _Float16* __restrict__ dprem, _Float16* __restrict__ dgm,
                         long n) {
  long i = (long)blockIdx.x * blockDim.x + threadIdx.x;
  if (i >= n) return;
  float gt = gate[i], gg = g[i], dp = dhp[i];  // dhp = kv@w1 (dhid = -dhp)
  float sg = 1.f / (1.f + __expf(-gt));
  float hs = gt * sg;                           // silu(gate)
  hid[i] = (_Float16)(hs * gg);                 // hidden
  dgm[i] = (_Float16)(dp * hs);                 // -dg
  float dsil = sg + gt * sg * (1.f - sg);       // silu'(gate)
  dprem[i] = (_Float16)((dp * gg) * dsil);      // -dpre
}

__global__ void lact_ew2(const float* __restrict__ gq, const float* __restrict__ gq2,
                         _Float16* __restrict__ act, long n) {
  long i = (long)blockIdx.x * blockDim.x + threadIdx.x;
  if (i >= n) return;
  float a = gq[i];
  float s = 1.f / (1.f + __expf(-a));
  act[i] = (_Float16)(a * s * gq2[i]);
}

__global__ void cvt_f32_f16(const float* __restrict__ s, _Float16* __restrict__ d,
                            long n) {
  long i = (long)blockIdx.x * blockDim.x + threadIdx.x;
  if (i < n) d[i] = (_Float16)s[i];
}

// ---------------------------------------------------------------------------
// Host orchestration
// ---------------------------------------------------------------------------
extern "C" void kernel_launch(void* const* d_in, const int* in_sizes, int n_in,
                              void* d_out, int out_size, void* d_ws, size_t ws_size,
                              hipStream_t stream) {
  const float* x    = (const float*)d_in[0];
  const float* lnw  = (const float*)d_in[1];
  const float* lnb  = (const float*)d_in[2];
  const float* ipw  = (const float*)d_in[3];
  const float* ipb  = (const float*)d_in[4];
  const float* opw  = (const float*)d_in[5];
  const float* opb  = (const float*)d_in[6];
  const float* w0   = (const float*)d_in[7];
  const float* w1   = (const float*)d_in[8];
  const float* w2   = (const float*)d_in[9];
  float* out = (float*)d_out;

  // ---- workspace layout ----
  char* w = (char*)d_ws;
  auto take = [&](size_t bytes) { char* p = w; w += bytes; return p; };
  _Float16* xnh    = (_Float16*)take((size_t)SQ * HS * 2);
  _Float16* wih    = (_Float16*)take((size_t)3 * HS * HS * 2);
  _Float16* woh    = (_Float16*)take((size_t)HS * HS * 2);
  _Float16* w0h    = (_Float16*)take((size_t)NHL * DIN * DIN * 2);
  _Float16* w1h    = (_Float16*)take((size_t)NHL * DIN * DIN * 2);
  _Float16* w2h    = (_Float16*)take((size_t)NHL * DIN * DIN * 2);
  _Float16* qkvh   = (_Float16*)take((size_t)SQ * 3 * HS * 2);
  _Float16* aoh    = (_Float16*)take((size_t)SQ * HS * 2);
  float*    gateb  = (float*)take((size_t)NZ * PERZ * 4);
  float*    gb     = (float*)take((size_t)NZ * PERZ * 4);
  float*    dhpb   = (float*)take((size_t)NZ * PERZ * 4);
  _Float16* hidb   = (_Float16*)take((size_t)NZ * PERZ * 2);
  _Float16* dpremb = (_Float16*)take((size_t)NZ * PERZ * 2);
  _Float16* dgmb   = (_Float16*)take((size_t)NZ * PERZ * 2);
  _Float16* w0nh   = (_Float16*)take((size_t)NZ * DIN * DIN * 2);
  _Float16* w1nh   = (_Float16*)take((size_t)NZ * DIN * DIN * 2);
  _Float16* w2nh   = (_Float16*)take((size_t)NZ * DIN * DIN * 2);

  auto cvt = [&](const float* s, _Float16* d, long n) {
    cvt_f32_f16<<<(unsigned)((n + 255) / 256), 256, 0, stream>>>(s, d, n);
  };
  cvt(ipw, wih, (long)3 * HS * HS);
  cvt(opw, woh, (long)HS * HS);
  cvt(w0, w0h, (long)NHL * DIN * DIN);
  cvt(w1, w1h, (long)NHL * DIN * DIN);
  cvt(w2, w2h, (long)NHL * DIN * DIN);

  ln_kernel<<<SQ, 256, 0, stream>>>(x, lnw, lnb, xnh);

  GemmP p;
  auto clr = [&]() { memset(&p, 0, sizeof(p)); p.zdiv = 1; };

  // qkv = xn @ in_proj_w^T + b     [4096 x 3072]
  clr();
  p.A = xnh; p.lda = HS;
  p.B = wih; p.ldb = HS;
  p.bias = ipb;
  p.Dh = qkvh; p.lddh = 3 * HS;
  p.M = SQ; p.N = 3 * HS; p.K = HS;
  gemm_wmma<false, true><<<dim3(SQ / 64, 3 * HS / 64, 1), 128, 0, stream>>>(p);

  // attention
  flash_attn<<<(NHA * (SQ / 16)) / 4, 128, 0, stream>>>(qkvh, aoh, SQ);

  // attn_out = ao @ out_proj_w^T + b  -> d_out
  clr();
  p.A = aoh; p.lda = HS;
  p.B = woh; p.ldb = HS;
  p.bias = opb;
  p.Df = out; p.lddf = HS; p.accum = 0;
  p.M = SQ; p.N = HS; p.K = HS;
  gemm_wmma<false, true><<<dim3(SQ / 64, HS / 64, 1), 128, 0, stream>>>(p);

  // ---- LaCT fast-weight TTT, batched over z = chunk*4 + head ----
  const long AZN = (long)CHK * HS;   // chunk stride into xn
  const long AZH = DIN;              // head stride into xn
  const long SZN = (long)NHL * PERZ; // chunk stride into slabs
  const long SZH = PERZ;
  const long WZN = (long)NHL * DIN * DIN;
  const long WZH = (long)DIN * DIN;

  // gate = kv @ w0^T
  clr();
  p.A = xnh; p.lda = HS; p.Azn = AZN; p.Azh = AZH;
  p.B = w0h; p.ldb = DIN; p.Bzh = WZH;
  p.Df = gateb; p.lddf = DIN; p.Dfzn = SZN; p.Dfzh = SZH;
  p.M = CHK; p.N = DIN; p.K = DIN; p.zdiv = NHL;
  gemm_wmma<false, true><<<dim3(CHK / 64, DIN / 64, NZ), 128, 0, stream>>>(p);
  // g = kv @ w2^T
  p.B = w2h; p.Df = gb;
  gemm_wmma<false, true><<<dim3(CHK / 64, DIN / 64, NZ), 128, 0, stream>>>(p);
  // dhp = kv @ w1   (dhid = -dhp)
  p.B = w1h; p.Df = dhpb;
  gemm_wmma<false, false><<<dim3(CHK / 64, DIN / 64, NZ), 128, 0, stream>>>(p);

  long nz = (long)NZ * PERZ;
  lact_ew1<<<(unsigned)(nz / 256), 256, 0, stream>>>(gateb, gb, dhpb, hidb,
                                                     dpremb, dgmb, nz);

  // w1n = w1 + kv^T @ hidden
  clr();
  p.A = xnh; p.lda = HS; p.Azn = AZN; p.Azh = AZH;
  p.B = hidb; p.ldb = DIN; p.Bzn = SZN; p.Bzh = SZH;
  p.C = w1; p.ldc = DIN; p.Czh = WZH;
  p.Dh = w1nh; p.lddh = DIN; p.Dhzn = WZN; p.Dhzh = WZH;
  p.M = DIN; p.N = DIN; p.K = CHK; p.zdiv = NHL;
  gemm_wmma<true, false><<<dim3(DIN / 64, DIN / 64, NZ), 128, 0, stream>>>(p);
  // w0n = w0 + (-dpre)^T @ kv
  clr();
  p.A = dpremb; p.lda = DIN; p.Azn = SZN; p.Azh = SZH;
  p.B = xnh; p.ldb = HS; p.Bzn = AZN; p.Bzh = AZH;
  p.C = w0; p.ldc = DIN; p.Czh = WZH;
  p.Dh = w0nh; p.lddh = DIN; p.Dhzn = WZN; p.Dhzh = WZH;
  p.M = DIN; p.N = DIN; p.K = CHK; p.zdiv = NHL;
  gemm_wmma<true, false><<<dim3(DIN / 64, DIN / 64, NZ), 128, 0, stream>>>(p);
  // w2n = w2 + (-dg)^T @ kv
  p.A = dgmb; p.C = w2; p.Dh = w2nh;
  gemm_wmma<true, false><<<dim3(DIN / 64, DIN / 64, NZ), 128, 0, stream>>>(p);

  // gq = c @ w0n^T ; gq2 = c @ w2n^T   (reuse gate/g slabs)
  clr();
  p.A = xnh; p.lda = HS; p.Azn = AZN; p.Azh = AZH;
  p.B = w0nh; p.ldb = DIN; p.Bzn = WZN; p.Bzh = WZH;
  p.Df = gateb; p.lddf = DIN; p.Dfzn = SZN; p.Dfzh = SZH;
  p.M = CHK; p.N = DIN; p.K = DIN; p.zdiv = NHL;
  gemm_wmma<false, true><<<dim3(CHK / 64, DIN / 64, NZ), 128, 0, stream>>>(p);
  p.B = w2nh; p.Df = gb;
  gemm_wmma<false, true><<<dim3(CHK / 64, DIN / 64, NZ), 128, 0, stream>>>(p);

  // act = silu(gq) * gq2  (reuse hid slab)
  lact_ew2<<<(unsigned)(nz / 256), 256, 0, stream>>>(gateb, gb, hidb, nz);

  // d_out += act @ w1n^T   (scattered back to [S, HS] layout)
  clr();
  p.A = hidb; p.lda = DIN; p.Azn = SZN; p.Azh = SZH;
  p.B = w1nh; p.ldb = DIN; p.Bzn = WZN; p.Bzh = WZH;
  p.Df = out; p.lddf = HS; p.Dfzn = (long)CHK * HS; p.Dfzh = DIN; p.accum = 1;
  p.M = CHK; p.N = DIN; p.K = DIN; p.zdiv = NHL;
  gemm_wmma<false, true><<<dim3(CHK / 64, DIN / 64, NZ), 128, 0, stream>>>(p);

  (void)in_sizes; (void)n_in; (void)out_size; (void)ws_size;
}
